// BiMambaEncoder_10806137716809
// MI455X (gfx1250) — compile-verified, matrix operand order checked by reading, not run on hardware
//
#include <hip/hip_runtime.h>
#include <hip/hip_bf16.h>

// ---------------------------------------------------------------------------
// BiMamba encoder for MI455X (gfx1250, wave32, WMMA).
// B=32, L=1024, D_MODEL=256, D_INNER=512, D_STATE=16, DT_RANK=16, NL=4.
// ---------------------------------------------------------------------------

#define MTOK   (32 * 1024)     // B*L tokens
#define DMODEL 256
#define DINNER 512
#define DSTATE 16
#define DTRANK 16
#define LSEQ   1024
#define TCHUNK 64              // scan time-chunk staged in LDS

typedef __attribute__((ext_vector_type(16))) _Float16 v16h;
typedef __attribute__((ext_vector_type(8)))  _Float16 v8h;
typedef __attribute__((ext_vector_type(8)))  float    v8f;

// -------------------------- f32 -> f16 convert -----------------------------
__global__ void k_f32_to_f16(const float* __restrict__ in, _Float16* __restrict__ out, int n) {
    int i = blockIdx.x * 256 + threadIdx.x;
    if (i < n) out[i] = (_Float16)in[i];
}

// ------------------------------ tokenizer ----------------------------------
// wave-per-token: build 136-wide concat in LDS, 136->256 linear, LayerNorm.
__global__ void k_tokenize(const float* __restrict__ X,
                           const float* __restrict__ embP, const float* __restrict__ embF,
                           const float* __restrict__ embD,
                           const float* __restrict__ lenW, const float* __restrict__ lenB,
                           const float* __restrict__ iatW, const float* __restrict__ iatB,
                           const float* __restrict__ fusW, const float* __restrict__ fusB,
                           const float* __restrict__ lnW,  const float* __restrict__ lnB,
                           float* __restrict__ feat, _Float16* __restrict__ featH) {
    __shared__ float sc[8][136];
    int wave = threadIdx.x >> 5;
    int lane = threadIdx.x & 31;
    int token = blockIdx.x * 8 + wave;
    const float* xr = X + (size_t)token * 5;
    float x0 = xr[0], x1 = xr[1], x2 = xr[2], x3 = xr[3], x4 = xr[4];
    int proto = (int)fminf(fmaxf(x0, 0.f), 255.f);
    int flags = (int)fminf(fmaxf(x2, 0.f), 63.f);
    int dirc  = (int)fminf(fmaxf(x4, 0.f), 1.f);
    for (int idx = lane; idx < 136; idx += 32) {
        float v;
        if      (idx < 32)  v = embP[proto * 32 + idx];
        else if (idx < 64)  { int j = idx - 32; v = x1 * lenW[j] + lenB[j]; }
        else if (idx < 96)  v = embF[flags * 32 + (idx - 64)];
        else if (idx < 128) { int j = idx - 96; v = x3 * iatW[j] + iatB[j]; }
        else                v = embD[dirc * 8 + (idx - 128)];
        sc[wave][idx] = v;
    }
    __syncthreads();
    float out[8];
    float s1 = 0.f, s2 = 0.f;
    #pragma unroll
    for (int j = 0; j < 8; ++j) {
        int o = lane + 32 * j;
        float acc = fusB[o];
        const float* wr = fusW + (size_t)o * 136;
        for (int k = 0; k < 136; ++k) acc += sc[wave][k] * wr[k];
        out[j] = acc; s1 += acc; s2 += acc * acc;
    }
    for (int m = 16; m >= 1; m >>= 1) { s1 += __shfl_xor(s1, m, 32); s2 += __shfl_xor(s2, m, 32); }
    float mean = s1 * (1.f / 256.f);
    float var  = s2 * (1.f / 256.f) - mean * mean;
    float inv  = rsqrtf(var + 1e-5f);
    #pragma unroll
    for (int j = 0; j < 8; ++j) {
        int o = lane + 32 * j;
        float v = (out[j] - mean) * inv * lnW[o] + lnB[o];
        feat [(size_t)token * DMODEL + o] = v;
        featH[(size_t)token * DMODEL + o] = (_Float16)v;
    }
}

// --------------------------- WMMA GEMM kernel ------------------------------
// C[M,N](f32) = A[M,K](f16) * W[N,K]^T(f16).  One wave per block; computes a
// 16-row stripe x NT 16-col tiles, reusing the A fragment across tiles.
// A-fragment layout (16-bit A 16x32): lane l holds row l%16; halves [0..7] are
// K = k0 + (l/16)*8 + i, halves [8..15] are K = k0 + 16 + (l/16)*8 + i.
// B is fed with the symmetric per-column layout (W row n = B column n).
template <int NT>
__global__ void k_gemm_wmma(const _Float16* __restrict__ A, const _Float16* __restrict__ W,
                            float* __restrict__ C, int K, int N) {
    int lane = threadIdx.x;
    int m0 = blockIdx.x * 16;
    int n0 = blockIdx.y * (16 * NT);
    int r16 = lane & 15;
    int half = lane >> 4;
    const _Float16* arow = A + (size_t)(m0 + r16) * K + half * 8;
    v8f acc[NT];
    #pragma unroll
    for (int i = 0; i < NT; ++i)
        #pragma unroll
        for (int j = 0; j < 8; ++j) acc[i][j] = 0.f;

    for (int k0 = 0; k0 < K; k0 += 32) {
        v8h alo = *(const v8h*)(arow + k0);
        v8h ahi = *(const v8h*)(arow + k0 + 16);
        v16h a;
        #pragma unroll
        for (int i = 0; i < 8; ++i) { a[i] = alo[i]; a[i + 8] = ahi[i]; }
        #pragma unroll
        for (int nt = 0; nt < NT; ++nt) {
            const _Float16* wrow = W + (size_t)(n0 + nt * 16 + r16) * K + half * 8 + k0;
            v8h blo = *(const v8h*)(wrow);
            v8h bhi = *(const v8h*)(wrow + 16);
            v16h bb;
            #pragma unroll
            for (int i = 0; i < 8; ++i) { bb[i] = blo[i]; bb[i + 8] = bhi[i]; }
            acc[nt] = __builtin_amdgcn_wmma_f32_16x16x32_f16(
                false, a, false, bb, (short)0, acc[nt], false, false);
        }
    }
    // C/D layout: VGPR r -> row m0 + (lane/16)*8 + r, col = n-tile + lane%16
    #pragma unroll
    for (int nt = 0; nt < NT; ++nt) {
        int col = n0 + nt * 16 + r16;
        #pragma unroll
        for (int r = 0; r < 8; ++r)
            C[(size_t)(m0 + half * 8 + r) * N + col] = acc[nt][r];
    }
}

// ---------------------- causal depthwise conv + silu -----------------------
// fwd: xc[t] = silu(b + sum_j w[j]*xi[t-3+j]);  rev: taps mirrored in time.
__global__ void k_conv(const float* __restrict__ xz, const float* __restrict__ cw,
                       const float* __restrict__ cb, float* __restrict__ xc,
                       _Float16* __restrict__ xcH, int rev) {
    int gid = blockIdx.x * 256 + threadIdx.x;     // over MTOK*DINNER
    int d = gid & (DINNER - 1);
    int token = gid >> 9;
    int t = token & (LSEQ - 1);
    const float* base = xz + (size_t)token * (2 * DINNER) + d;   // xi half of xz
    float w0 = cw[d * 4 + 0], w1 = cw[d * 4 + 1], w2 = cw[d * 4 + 2], w3 = cw[d * 4 + 3];
    float acc = cb[d];
    const int S = 2 * DINNER;                      // time stride in floats
    if (!rev) {
        if (t >= 3) acc += w0 * base[-3 * S];
        if (t >= 2) acc += w1 * base[-2 * S];
        if (t >= 1) acc += w2 * base[-1 * S];
        acc += w3 * base[0];
    } else {
        acc += w3 * base[0];
        if (t <= LSEQ - 2) acc += w2 * base[1 * S];
        if (t <= LSEQ - 3) acc += w1 * base[2 * S];
        if (t <= LSEQ - 4) acc += w0 * base[3 * S];
    }
    float v = acc / (1.f + expf(-acc));
    xc[gid] = v;
    xcH[gid] = (_Float16)v;
}

// ---------------------------- selective scan -------------------------------
// block = (batch b, 256-channel chunk); thread owns one channel d with h[16]
// in registers; dt-projection (K=16) fused; gating + silu(z) fused; stages
// 64 steps of the shared (dt_rank,B,C) 48-vector in LDS per barrier round.
__global__ void k_scan(const float* __restrict__ dtbc, const float* __restrict__ xc,
                       const float* __restrict__ xz, const float* __restrict__ dtW,
                       const float* __restrict__ dtB, const float* __restrict__ Alog,
                       const float* __restrict__ Dp, _Float16* __restrict__ yH, int rev) {
    __shared__ float sb[TCHUNK][48];
    int b = blockIdx.y;
    int d = blockIdx.x * 256 + threadIdx.x;
    float a[DSTATE], h[DSTATE], wv[DTRANK];
    #pragma unroll
    for (int n = 0; n < DSTATE; ++n) {
        a[n] = -expf(Alog[d * DSTATE + n]);
        h[n] = 0.f;
        wv[n] = dtW[d * DTRANK + n];
    }
    float bias = dtB[d], dd = Dp[d];
    size_t tokBase = (size_t)b * LSEQ;

    for (int ci = 0; ci < LSEQ / TCHUNK; ++ci) {
        int c = rev ? (LSEQ / TCHUNK - 1 - ci) : ci;
        for (int i = threadIdx.x; i < TCHUNK * 48; i += 256)
            sb[i / 48][i % 48] = dtbc[(tokBase + (size_t)c * TCHUNK + i / 48) * 48 + (i % 48)];
        __syncthreads();
        for (int tt0 = 0; tt0 < TCHUNK; ++tt0) {
            int tt = rev ? (TCHUNK - 1 - tt0) : tt0;
            size_t tok = tokBase + (size_t)c * TCHUNK + tt;
            float dtv = bias;
            #pragma unroll
            for (int k = 0; k < DTRANK; ++k) dtv += sb[tt][k] * wv[k];
            dtv = (dtv > 20.f) ? dtv : log1pf(expf(dtv));            // softplus
            float u = xc[tok * DINNER + d];
            float z = xz[tok * (2 * DINNER) + DINNER + d];
            float du = dtv * u;
            float ys = 0.f;
            #pragma unroll
            for (int n = 0; n < DSTATE; ++n) {
                float dA = expf(dtv * a[n]);
                h[n] = dA * h[n] + du * sb[tt][16 + n];
                ys += h[n] * sb[tt][32 + n];
            }
            float y = (ys + u * dd) * (z / (1.f + expf(-z)));        // * silu(z)
            yH[tok * DINNER + d] = (_Float16)y;
        }
        __syncthreads();
    }
}

// -------------------- combine directions + residual + LN -------------------
__global__ void k_combine(const float* __restrict__ of, const float* __restrict__ orv,
                          float* __restrict__ feat, _Float16* __restrict__ featH,
                          const float* __restrict__ nw, const float* __restrict__ nb) {
    int wave = threadIdx.x >> 5, lane = threadIdx.x & 31;
    size_t token = (size_t)blockIdx.x * 8 + wave;
    float v[8], s1 = 0.f, s2 = 0.f;
    #pragma unroll
    for (int j = 0; j < 8; ++j) {
        size_t idx = token * DMODEL + lane + 32 * j;
        float x = 0.5f * (of[idx] + orv[idx]) + feat[idx];
        v[j] = x; s1 += x; s2 += x * x;
    }
    for (int m = 16; m >= 1; m >>= 1) { s1 += __shfl_xor(s1, m, 32); s2 += __shfl_xor(s2, m, 32); }
    float mean = s1 * (1.f / 256.f);
    float var  = s2 * (1.f / 256.f) - mean * mean;
    float inv  = rsqrtf(var + 1e-5f);
    #pragma unroll
    for (int j = 0; j < 8; ++j) {
        size_t idx = token * DMODEL + lane + 32 * j;
        int o = lane + 32 * j;
        float x = (v[j] - mean) * inv * nw[o] + nb[o];
        feat[idx] = x;
        featH[idx] = (_Float16)x;
    }
}

// ------------------------------- launcher ----------------------------------
extern "C" void kernel_launch(void* const* d_in, const int* in_sizes, int n_in,
                              void* d_out, int out_size, void* d_ws, size_t ws_size,
                              hipStream_t stream) {
    auto in = [&](int i) { return (const float*)d_in[i]; };
    // input layout (setup_inputs insertion order):
    // 0:x  1:emb_proto 2:emb_flags 3:emb_dir 4:len_w 5:len_b 6:iat_w 7:iat_b
    // 8:fus_w 9:fus_b 10:ln_w 11:ln_b
    // 12+9i .. : mambas_fwd[i] {in_w,conv_w,conv_b,xproj_w,dt_w,dt_b,A_log,D,out_w}
    // 48+9i .. : mambas_rev[i]
    // 84:norm_w 85:norm_b
    (void)in_sizes; (void)n_in; (void)out_size; (void)ws_size;

    uint8_t* ws = (uint8_t*)d_ws;
    size_t off = 0;
    auto bump = [&](size_t bytes) { size_t o = off; off += (bytes + 255) & ~(size_t)255; return o; };
    float*    feat  = (float*)   (ws + bump((size_t)MTOK * DMODEL * 4));
    _Float16* featH = (_Float16*)(ws + bump((size_t)MTOK * DMODEL * 2));
    float*    xz    = (float*)   (ws + bump((size_t)MTOK * 2 * DINNER * 4));
    float*    xc    = (float*)   (ws + bump((size_t)MTOK * DINNER * 4));
    _Float16* xcH   = (_Float16*)(ws + bump((size_t)MTOK * DINNER * 2));
    float*    dtbc  = (float*)   (ws + bump((size_t)MTOK * 48 * 4));
    _Float16* yH    = (_Float16*)(ws + bump((size_t)MTOK * DINNER * 2));
    float*    outF  = (float*)   (ws + bump((size_t)MTOK * DMODEL * 4));
    float*    outR  = (float*)   (ws + bump((size_t)MTOK * DMODEL * 4));
    const size_t WSLOT = 262144 + 24576 + 131072;  // in_w + xproj_w + out_w halves
    _Float16* wH    = (_Float16*)(ws + bump(8 * WSLOT * 2));

    // ---- convert weights to f16 (deterministic, every call) ----
    for (int s = 0; s < 8; ++s) {
        int layer = s >> 1;
        int b0 = ((s & 1) ? 48 : 12) + 9 * layer;
        _Float16* wi = wH + (size_t)s * WSLOT;
        k_f32_to_f16<<<(262144 + 255) / 256, 256, 0, stream>>>(in(b0 + 0), wi, 262144);
        k_f32_to_f16<<<(24576 + 255) / 256, 256, 0, stream>>>(in(b0 + 3), wi + 262144, 24576);
        k_f32_to_f16<<<(131072 + 255) / 256, 256, 0, stream>>>(in(b0 + 8), wi + 262144 + 24576, 131072);
    }

    // ---- tokenize ----
    k_tokenize<<<MTOK / 8, 256, 0, stream>>>(in(0), in(1), in(2), in(3), in(4), in(5),
                                             in(6), in(7), in(8), in(9), in(10), in(11),
                                             feat, featH);

    // ---- layers ----
    for (int layer = 0; layer < 4; ++layer) {
        for (int dir = 0; dir < 2; ++dir) {
            int s = layer * 2 + dir;
            int b0 = (dir ? 48 : 12) + 9 * layer;
            _Float16* wIn = wH + (size_t)s * WSLOT;
            _Float16* wX  = wIn + 262144;
            _Float16* wO  = wIn + 262144 + 24576;
            float* outBuf = dir ? outR : outF;

            // in_proj: [MTOK,256] x [1024,256]^T -> xz [MTOK,1024]
            k_gemm_wmma<4><<<dim3(MTOK / 16, (2 * DINNER) / 64), 32, 0, stream>>>(
                featH, wIn, xz, DMODEL, 2 * DINNER);
            // depthwise causal conv + silu (rev flips taps)
            k_conv<<<(MTOK * DINNER) / 256, 256, 0, stream>>>(
                xz, in(b0 + 1), in(b0 + 2), xc, xcH, dir);
            // x_proj: [MTOK,512] x [48,512]^T -> dtbc [MTOK,48]
            k_gemm_wmma<3><<<dim3(MTOK / 16, 1), 32, 0, stream>>>(
                xcH, wX, dtbc, DINNER, 48);
            // selective scan (dt proj + softplus + gating fused), rev flips time
            k_scan<<<dim3(DINNER / 256, 32), 256, 0, stream>>>(
                dtbc, xc, xz, in(b0 + 4), in(b0 + 5), in(b0 + 6), in(b0 + 7), yH, dir);
            // out_proj: [MTOK,512] x [256,512]^T -> out [MTOK,256]
            k_gemm_wmma<4><<<dim3(MTOK / 16, DMODEL / 64), 32, 0, stream>>>(
                yH, wO, outBuf, DINNER, DMODEL);
        }
        k_combine<<<MTOK / 8, 256, 0, stream>>>(outF, outR, feat, featH, in(84), in(85));
    }

    hipMemcpyAsync(d_out, feat, (size_t)MTOK * DMODEL * sizeof(float),
                   hipMemcpyDeviceToDevice, stream);
}
